// enHOPE_82274393522264
// MI455X (gfx1250) — matrix operand order repaired and checked by reading, not compiled, not run on hardware
//
#include <hip/hip_runtime.h>
#include <math.h>

// ---------------------------------------------------------------------------
// MI455X / gfx1250: 3-GEMM MLP chain on v_wmma_f32_16x16x32_bf16 with
// double-buffered GLOBAL_LOAD_ASYNC_TO_LDS staging (ASYNCcnt pipeline).
//   pe = (x1 @ C)^2 ; h = sigmoid(pe @ w + b) ; out = h @ fc2_w^T
// Block tile 128x256, 8 waves (2Mx4N), wave tile 64x64 (4x4 WMMA) to cut
// LDS traffic per WMMA (ds_load_b128 : wmma = 1 : 1).
// ---------------------------------------------------------------------------

typedef __bf16 bf16_t;
typedef __attribute__((ext_vector_type(16))) __bf16 v16bf;
typedef __attribute__((ext_vector_type(8)))  __bf16 v8bf;
typedef __attribute__((ext_vector_type(8)))  float  v8f;
typedef __attribute__((ext_vector_type(4)))  int    v4i;

union FragAB { v16bf full; v8bf h[2]; };

#define AS1 __attribute__((address_space(1)))
#define AS3 __attribute__((address_space(3)))

// ---- async global->LDS copy (16B per lane), ASYNCcnt-tracked ----
__device__ __forceinline__ void async_cp16(const bf16_t* g, bf16_t* l) {
#if __has_builtin(__builtin_amdgcn_global_load_async_to_lds_b128)
  __builtin_amdgcn_global_load_async_to_lds_b128(
      (AS1 v4i*)(void*)g, (AS3 v4i*)(void*)l, 0, 0);
#else
  unsigned lds_addr = (unsigned)(unsigned long long)(AS3 void*)l;
  asm volatile("global_load_async_to_lds_b128 %0, %1, off"
               :: "v"(lds_addr), "v"(g) : "memory");
#endif
}

template <int N>
__device__ __forceinline__ void wait_async() {
#if __has_builtin(__builtin_amdgcn_s_wait_asynccnt)
  __builtin_amdgcn_s_wait_asynccnt((unsigned short)N);
#else
  asm volatile("s_wait_asynccnt %0" :: "n"(N) : "memory");
#endif
}

// ---------------------------------------------------------------------------
// Prep kernels
// ---------------------------------------------------------------------------

__global__ void enhope_build_x1(const float* __restrict__ x, bf16_t* __restrict__ x1,
                                int B, int D, int Kp) {
  int idx = blockIdx.x * blockDim.x + threadIdx.x;
  int total = B * Kp;
  if (idx >= total) return;
  int row = idx / Kp;
  int c   = idx - row * Kp;
  float v = (c < D) ? x[(size_t)row * D + c] : (c == D ? 1.0f : 0.0f);
  x1[idx] = (bf16_t)v;
}

// dst[c, r] = (bf16) src[r, c]; dst is [Ccols x dstStride] row-major,
// rows r in [R, dstStride) are zero-padded.  block (32, 8).
__global__ void enhope_transpose_bf16(const float* __restrict__ src, bf16_t* __restrict__ dst,
                                      int R, int Ccols, int dstStride) {
  __shared__ float tile[32][33];
  int c0 = blockIdx.x * 32;
  int r0 = blockIdx.y * 32;
  int tx = threadIdx.x, ty = threadIdx.y;
#pragma unroll
  for (int i = 0; i < 32; i += 8) {
    int r = r0 + ty + i, c = c0 + tx;
    float v = 0.0f;
    if (r < R && c < Ccols) v = src[(size_t)r * Ccols + c];
    tile[ty + i][tx] = v;
  }
  __syncthreads();
#pragma unroll
  for (int i = 0; i < 32; i += 8) {
    int dr = c0 + ty + i;
    int dc = r0 + tx;
    if (dr < Ccols && dc < dstStride)
      dst[(size_t)dr * dstStride + dc] = (bf16_t)tile[tx][ty + i];
  }
}

__global__ void enhope_convert_bf16(const float* __restrict__ src, bf16_t* __restrict__ dst, int n) {
  int i = blockIdx.x * blockDim.x + threadIdx.x;
  if (i < n) dst[i] = (bf16_t)src[i];
}

// ---------------------------------------------------------------------------
// WMMA GEMM: D = A(bf16,[M x K]) * Bt(bf16,[N x K])^T, f32 accumulate.
// EPI 0: out = bf16(acc^2) | EPI 1: out = bf16(sigmoid(acc+b)) | EPI 2: f32.
// Block tile 128x256, BK=32, 8 waves (2Mx4N), wave tile 64x64 (4x4 WMMA).
// Double-buffered LDS filled by the async-to-LDS engine (6 b128/thread/stage).
// ---------------------------------------------------------------------------
template <int EPI>
__global__ __launch_bounds__(256) void enhope_gemm_bf16(
    const bf16_t* __restrict__ A, int lda,
    const bf16_t* __restrict__ Bt, int ldb,
    float* __restrict__ outF, bf16_t* __restrict__ outB,
    const float* __restrict__ bias, int K, int ldo) {
  constexpr int BM = 128, BN = 256, BK = 32;
  constexpr int LDST = 40;  // bf16 elems per LDS row (32 + 8 pad -> 80B stride)

  __shared__ bf16_t As[2][BM * LDST];   // 2 x 10240 B
  __shared__ bf16_t Bs[2][BN * LDST];   // 2 x 20480 B   (total 60 KB)

  const int tid  = threadIdx.x;
  const int lane = tid & 31;
  const int wave = tid >> 5;
  const int wm   = wave >> 2;   // 0..1 : M direction
  const int wn   = wave & 3;    // 0..3 : N direction
  const int r    = lane & 15;
  const int hl   = lane >> 4;   // lane half

  const int bm = blockIdx.y * BM;
  const int bn = blockIdx.x * BN;

  // staging plan: 16B chunks; thread owns column cCol of rows cRow+{0,64,...}
  const int cRow = tid >> 2;          // 0..63
  const int cCol = (tid & 3) * 8;     // bf16 offset of 16B chunk

  v8f acc[4][4] = {};

  // issue 6 async b128 copies for one K-stage into LDS buffer `buf`
  auto stage = [&](int buf, int k0) {
    const bf16_t* gA = A + (size_t)(bm + cRow) * lda + k0 + cCol;
    async_cp16(gA,                     &As[buf][cRow * LDST + cCol]);
    async_cp16(gA + (size_t)64 * lda,  &As[buf][(cRow + 64) * LDST + cCol]);
    const bf16_t* gB = Bt + (size_t)(bn + cRow) * ldb + k0 + cCol;
    async_cp16(gB,                     &Bs[buf][cRow * LDST + cCol]);
    async_cp16(gB + (size_t)64 * ldb,  &Bs[buf][(cRow + 64) * LDST + cCol]);
    async_cp16(gB + (size_t)128 * ldb, &Bs[buf][(cRow + 128) * LDST + cCol]);
    async_cp16(gB + (size_t)192 * ldb, &Bs[buf][(cRow + 192) * LDST + cCol]);
  };

  const int nk = K / BK;
  stage(0, 0);

  for (int s = 0; s < nk; ++s) {
    const int cur = s & 1;
    if (s + 1 < nk) {
      stage(cur ^ 1, (s + 1) * BK);   // overlap next fill with this compute
      wait_async<6>();                // oldest 6 (== buffer `cur`) complete
    } else {
      wait_async<0>();
    }
    __syncthreads();                  // everyone's fill of `cur` visible

    FragAB afrag[4], bfrag[4];
#pragma unroll
    for (int mt = 0; mt < 4; ++mt) {
      // A 16x32: lane half 0 -> K 0..7 / 16..23, half 1 -> K 8..15 / 24..31
      const bf16_t* p = &As[cur][(wm * 64 + mt * 16 + r) * LDST + hl * 8];
      afrag[mt].h[0] = *(const v8bf*)p;
      afrag[mt].h[1] = *(const v8bf*)(p + 16);
    }
#pragma unroll
    for (int nt = 0; nt < 4; ++nt) {
      // B 32x16 (as Bt rows): lane half 0 -> K 0..7 / 16..23, half 1 -> K 8..15 / 24..31
      const bf16_t* p = &Bs[cur][(wn * 64 + nt * 16 + r) * LDST + hl * 8];
      bfrag[nt].h[0] = *(const v8bf*)p;
      bfrag[nt].h[1] = *(const v8bf*)(p + 16);
    }

#pragma unroll
    for (int mt = 0; mt < 4; ++mt)
#pragma unroll
      for (int nt = 0; nt < 4; ++nt)
        acc[mt][nt] = __builtin_amdgcn_wmma_f32_16x16x32_bf16(
            false, afrag[mt].full, false, bfrag[nt].full,
            (short)0, acc[mt][nt], false, false);

    __syncthreads();  // all ds_loads of `cur` done before it is re-filled
  }

  // ---- fused epilogue; C/D layout: row = base + hl*8 + i, col = base + r ----
#pragma unroll
  for (int mt = 0; mt < 4; ++mt) {
#pragma unroll
    for (int nt = 0; nt < 4; ++nt) {
      const int col  = bn + wn * 64 + nt * 16 + r;
      const int row0 = bm + wm * 64 + mt * 16 + hl * 8;
#pragma unroll
      for (int i = 0; i < 8; ++i) {
        float v = acc[mt][nt][i];
        size_t off = (size_t)(row0 + i) * ldo + col;
        if (EPI == 0) {
          outB[off] = (bf16_t)(v * v);
        } else if (EPI == 1) {
          float sgm = 1.0f / (1.0f + __expf(-(v + bias[col])));
          outB[off] = (bf16_t)sgm;
        } else {
          outF[off] = v;
        }
      }
    }
  }
}

// ---------------------------------------------------------------------------
// Host launcher
// ---------------------------------------------------------------------------
extern "C" void kernel_launch(void* const* d_in, const int* in_sizes, int n_in,
                              void* d_out, int out_size, void* d_ws, size_t ws_size,
                              hipStream_t stream) {
  const float* x   = (const float*)d_in[0];  // [4096, 1024]
  const float* w   = (const float*)d_in[1];  // [4096, 2048]
  const float* b   = (const float*)d_in[2];  // [2048]
  const float* fc2 = (const float*)d_in[3];  // [512, 2048]
  const float* C   = (const float*)d_in[4];  // [1025, 4096]

  const int B = 4096, D = 1024, Kp = 1056, F = 4096, H = 2048, O = 512;

  char* ws = (char*)d_ws;
  bf16_t* x1  = (bf16_t*)ws; ws += (size_t)B * Kp * sizeof(bf16_t);  // 8.65 MB
  bf16_t* Ct  = (bf16_t*)ws; ws += (size_t)F * Kp * sizeof(bf16_t);  // 8.65 MB
  bf16_t* wT  = (bf16_t*)ws; ws += (size_t)H * F  * sizeof(bf16_t);  // 16.8 MB
  bf16_t* f2  = (bf16_t*)ws; ws += (size_t)O * H  * sizeof(bf16_t);  // 2.1 MB
  bf16_t* pe  = (bf16_t*)ws; ws += (size_t)B * F  * sizeof(bf16_t);  // 33.6 MB
  bf16_t* hh  = (bf16_t*)ws;                                          // 16.8 MB

  enhope_build_x1<<<(B * Kp + 255) / 256, 256, 0, stream>>>(x, x1, B, D, Kp);
  enhope_transpose_bf16<<<dim3(F / 32, Kp / 32), dim3(32, 8), 0, stream>>>(C, Ct, D + 1, F, Kp);
  enhope_transpose_bf16<<<dim3(H / 32, F / 32), dim3(32, 8), 0, stream>>>(w, wT, F, H, F);
  enhope_convert_bf16<<<(O * H + 255) / 256, 256, 0, stream>>>(fc2, f2, O * H);

  // GEMM 1: pe = bf16( (x1 @ C)^2 )      [B x F], K = Kp
  enhope_gemm_bf16<0><<<dim3(F / 256, B / 128), 256, 0, stream>>>(
      x1, Kp, Ct, Kp, nullptr, pe, nullptr, Kp, F);
  // GEMM 2: h = bf16( sigmoid(pe @ w + b) )  [B x H], K = F
  enhope_gemm_bf16<1><<<dim3(H / 256, B / 128), 256, 0, stream>>>(
      pe, F, wT, F, nullptr, hh, b, F, H);
  // GEMM 3: out = f32( h @ fc2^T )       [B x O], K = H
  enhope_gemm_bf16<2><<<dim3(O / 256, B / 128), 256, 0, stream>>>(
      hh, H, f2, H, (float*)d_out, nullptr, nullptr, H, O);
}
